// LabelCrossAttention_57492432224483
// MI455X (gfx1250) — matrix operand order repaired and checked by reading, not compiled, not run on hardware
//
#include <hip/hip_runtime.h>
#include <hip/hip_bf16.h>

#define BB 4
#define SS 4096
#define DD 768
#define CC 8000

typedef __attribute__((ext_vector_type(16))) _Float16 v16h;
typedef __attribute__((ext_vector_type(8)))  _Float16 v8h;
typedef __attribute__((ext_vector_type(4)))  _Float16 v4h;
typedef __attribute__((ext_vector_type(8)))  float    v8f;

union Frag16 { v16h v; v8h h[2]; };

constexpr int    SC_STRIDE  = SS + 8;                         // 4104 halves
constexpr int    Q_STRIDE   = DD + 8;                         // 776 halves
constexpr size_t SC_BYTES   = (size_t)16 * SC_STRIDE * 2;     // 131328
constexpr size_t Q_BYTES    = (size_t)16 * Q_STRIDE * 2;      // 24832
constexpr size_t F_OFF      = SC_BYTES + Q_BYTES;
constexpr size_t SMEM_BYTES = F_OFF + (size_t)(128 * 4 + 16 * 5) * sizeof(float); // 158528

__device__ __forceinline__ float halfmax16(float v) {
  v = fmaxf(v, __shfl_xor(v, 1, 32));
  v = fmaxf(v, __shfl_xor(v, 2, 32));
  v = fmaxf(v, __shfl_xor(v, 4, 32));
  v = fmaxf(v, __shfl_xor(v, 8, 32));
  return v;
}
__device__ __forceinline__ float halfsum16(float v) {
  v += __shfl_xor(v, 1, 32);
  v += __shfl_xor(v, 2, 32);
  v += __shfl_xor(v, 4, 32);
  v += __shfl_xor(v, 8, 32);
  return v;
}

// ---------------------------------------------------------------------------
// Kernel 0: one-shot f32 -> f16 conversion of Wk / Wv
// ---------------------------------------------------------------------------
__global__ void __launch_bounds__(256) wcvt_kernel(
    const float* __restrict__ Wk, const float* __restrict__ Wv,
    _Float16* __restrict__ WhK, _Float16* __restrict__ WhV)
{
  const int i = (blockIdx.x * 256 + threadIdx.x) * 4;   // DD*DD = 589824 = 576*256*4
  float4 a = *(const float4*)(Wk + i);
  float4 b = *(const float4*)(Wv + i);
  v4h ha = { (_Float16)a.x, (_Float16)a.y, (_Float16)a.z, (_Float16)a.w };
  v4h hb = { (_Float16)b.x, (_Float16)b.y, (_Float16)b.z, (_Float16)b.w };
  *(v4h*)(WhK + i) = ha;
  *(v4h*)(WhV + i) = hb;
}

// ---------------------------------------------------------------------------
// Kernel A: K = x @ Wk^T (f16 [b,s,d]); Vt = (x @ Wv^T + bv)^T (f16 [b,d,s]).
// Waves 0-3: K, waves 4-7: V. Software-pipelined: weight fragments for step
// e+32 are loaded while the WMMAs for step e execute.
// ---------------------------------------------------------------------------
__global__ void __launch_bounds__(256) kv_proj_kernel(
    const float* __restrict__ x, const _Float16* __restrict__ WhK,
    const _Float16* __restrict__ WhV, const float* __restrict__ bv,
    _Float16* __restrict__ Kh, _Float16* __restrict__ Vt)
{
  __shared__ _Float16 xT[16 * Q_STRIDE];

  const int tid  = threadIdx.x;
  const int wave = tid >> 5, lane = tid & 31;
  const int lh = lane >> 4, l = lane & 15;
  const int mat = wave >> 2;                        // 0 = K, 1 = V
  const int wn  = wave & 3;
  const int m0  = blockIdx.x * 16;
  const int n0  = blockIdx.y * 256 + wn * 64;

  for (int i = tid; i < 16 * DD; i += 256) {
    int r = i / DD, d = i - r * DD;
    xT[r * Q_STRIDE + d] = (_Float16)x[(size_t)(m0 + r) * DD + d];
  }
  __syncthreads();

  const _Float16* W = mat ? WhV : WhK;
  const _Float16* ar = xT + l * Q_STRIDE + lh * 16;
  const _Float16* wr = W + (size_t)(n0 + l) * DD + lh * 16;

  auto loadA = [&](Frag16& A, int e) {
    A.h[0] = *(const v8h*)(ar + e);
    A.h[1] = *(const v8h*)(ar + e + 8);
  };
  auto loadB = [&](Frag16 (&Bf)[4], int e) {
#pragma unroll
    for (int t = 0; t < 4; ++t) {
      const _Float16* wp = wr + (size_t)(t * 16) * DD + e;
      Bf[t].h[0] = *(const v8h*)(wp);
      Bf[t].h[1] = *(const v8h*)(wp + 8);
    }
  };

  v8f acc[4] = {};
  Frag16 B0[4], B1[4];
  loadB(B0, 0);
#pragma unroll 3
  for (int e = 0; e < DD; e += 64) {
    Frag16 Aa, Ab;
    loadA(Aa, e);
    loadB(B1, e + 32);
#pragma unroll
    for (int t = 0; t < 4; ++t)
      acc[t] = __builtin_amdgcn_wmma_f32_16x16x32_f16(false, Aa.v, false, B0[t].v,
                                                      (short)0, acc[t], false, false);
    loadA(Ab, e + 32);
    loadB(B0, (e + 64 < DD) ? e + 64 : 0);
#pragma unroll
    for (int t = 0; t < 4; ++t)
      acc[t] = __builtin_amdgcn_wmma_f32_16x16x32_f16(false, Ab.v, false, B1[t].v,
                                                      (short)0, acc[t], false, false);
  }

  if (mat == 0) {
#pragma unroll
    for (int t = 0; t < 4; ++t) {
      const int n = n0 + t * 16 + l;
#pragma unroll
      for (int i = 0; i < 8; ++i) {
        const int m = m0 + i + lh * 8;
        Kh[(size_t)m * DD + n] = (_Float16)acc[t][i];
      }
    }
  } else {
#pragma unroll
    for (int t = 0; t < 4; ++t) {
      const int n = n0 + t * 16 + l;
      const float bvn = bv[n];
#pragma unroll
      for (int i = 0; i < 8; ++i) {
        const int m = m0 + i + lh * 8;
        const int bi = m >> 12, s = m & (SS - 1);
        Vt[((size_t)bi * DD + n) * SS + s] = (_Float16)(acc[t][i] + bvn);
      }
    }
  }
}

// ---------------------------------------------------------------------------
// Kernel B: per (batch, 16-label tile): scores -> softmax -> y -> LN -> dot
// ---------------------------------------------------------------------------
__global__ void __launch_bounds__(256) attn_kernel(
    const float* __restrict__ Q, const int* __restrict__ amask,
    const _Float16* __restrict__ Kh, const _Float16* __restrict__ Vt,
    const float* __restrict__ lng, const float* __restrict__ lnb,
    const float* __restrict__ wout, const float* __restrict__ bout,
    float* __restrict__ out)
{
  extern __shared__ char smem[];
  _Float16* scP = (_Float16*)smem;                  // [16][SC_STRIDE] scores->probs
  _Float16* qT  = (_Float16*)(smem + SC_BYTES);     // [16][Q_STRIDE]
  float* fbuf   = (float*)(smem + F_OFF);
  float* wmax   = fbuf;                             // [8][16]
  float* red1   = fbuf + 128;                       // [8][16]
  float* red2   = fbuf + 256;                       // [8][16]
  float* outred = fbuf + 384;                       // [8][16]
  float* mrow   = fbuf + 512;                       // [16]
  float* lsum   = fbuf + 528;                       // [16]
  float* linv   = fbuf + 544;                       // [16]
  float* muS    = fbuf + 560;                       // [16]
  float* rsS    = fbuf + 576;                       // [16]

  const int tid  = threadIdx.x;
  const int wave = tid >> 5, lane = tid & 31;
  const int lh = lane >> 4, l = lane & 15;
  const int c0 = blockIdx.x * 16;
  const int b  = blockIdx.y;

  for (int i = tid; i < 16 * DD; i += 256) {
    int r = i / DD, d = i - r * DD;
    qT[r * Q_STRIDE + d] = (_Float16)Q[(size_t)(c0 + r) * DD + d];
  }
  if (tid < 16) lsum[tid] = 0.f;
  __syncthreads();

  // ---- pass 1: scores = Q @ K^T, masked, track per-row max -----------------
  const _Float16* Kb = Kh + (size_t)b * SS * DD;
  const _Float16* qr = qT + l * Q_STRIDE + lh * 16;
  float rmax[8];
#pragma unroll
  for (int i = 0; i < 8; ++i) rmax[i] = -65000.f;

  for (int st = wave; st < SS / 16; st += 8) {
    const int s0 = st * 16;
    const _Float16* kr = Kb + (size_t)(s0 + l) * DD + lh * 16;

    auto loadQA = [&](Frag16& A, int e) {
      A.h[0] = *(const v8h*)(qr + e);
      A.h[1] = *(const v8h*)(qr + e + 8);
    };
    auto loadKB = [&](Frag16& Bk, int e) {
      Bk.h[0] = *(const v8h*)(kr + e);
      Bk.h[1] = *(const v8h*)(kr + e + 8);
    };

    v8f acc0 = {}, acc1 = {};                       // two chains -> WMMA ILP
    Frag16 B0, B1;
    loadKB(B0, 0);
#pragma unroll 3
    for (int e = 0; e < DD; e += 64) {
      Frag16 Aa, Ab;
      loadQA(Aa, e);
      loadKB(B1, e + 32);
      acc0 = __builtin_amdgcn_wmma_f32_16x16x32_f16(false, Aa.v, false, B0.v,
                                                    (short)0, acc0, false, false);
      loadQA(Ab, e + 32);
      loadKB(B0, (e + 64 < DD) ? e + 64 : 0);
      acc1 = __builtin_amdgcn_wmma_f32_16x16x32_f16(false, Ab.v, false, B1.v,
                                                    (short)0, acc1, false, false);
    }
    v8f acc = acc0 + acc1;
    const int mk = amask[b * SS + s0 + l];          // column mask (s = s0 + l)
#pragma unroll
    for (int i = 0; i < 8; ++i) {
      float v = mk ? acc[i] : -65000.f;
      rmax[i] = fmaxf(rmax[i], v);
      scP[(i + lh * 8) * SC_STRIDE + s0 + l] = (_Float16)v;
    }
  }
#pragma unroll
  for (int i = 0; i < 8; ++i) {
    float v = halfmax16(rmax[i]);
    if (l == 0) wmax[wave * 16 + i + lh * 8] = v;
  }
  __syncthreads();

  if (tid < 16) {
    float m = -65000.f;
    for (int w = 0; w < 8; ++w) m = fmaxf(m, wmax[w * 16 + tid]);
    mrow[tid] = fmaxf(m, -60000.f);
  }
  __syncthreads();

  // ---- pass 1.5: p = exp(s - m), row sums; overwrite scores with probs -----
  {
    const int r = tid >> 4, sub = tid & 15;
    const float m = mrow[r];
    float part = 0.f;
    for (int s = sub; s < SS; s += 16) {
      float p = __expf((float)scP[r * SC_STRIDE + s] - m);
      part += p;
      scP[r * SC_STRIDE + s] = (_Float16)p;
    }
    part = halfsum16(part);
    if (sub == 0) lsum[r] = part;
  }
  __syncthreads();
  if (tid < 16) linv[tid] = 1.f / (lsum[tid] + 1e-30f);
  __syncthreads();

  // ---- pass 2: y = P @ V (waves partition D; double-buffered fragments) ----
  const int n0 = wave * 96;
  const _Float16* Vb = Vt + (size_t)b * DD * SS + (size_t)(n0 + l) * SS + lh * 16;
  const _Float16* pr = scP + l * SC_STRIDE + lh * 16;

  auto loadPA = [&](Frag16& A, int s) {
    A.h[0] = *(const v8h*)(pr + s);
    A.h[1] = *(const v8h*)(pr + s + 8);
  };
  auto loadVB = [&](Frag16 (&Bv)[6], int s) {
#pragma unroll
    for (int t = 0; t < 6; ++t) {
      const _Float16* vr = Vb + (size_t)(t * 16) * SS + s;
      Bv[t].h[0] = *(const v8h*)(vr);
      Bv[t].h[1] = *(const v8h*)(vr + 8);
    }
  };

  v8f yacc[6] = {};
  Frag16 A0, A1, B0[6], B1[6];
  loadPA(A0, 0);
  loadVB(B0, 0);
#pragma unroll 1
  for (int s = 0; s < SS; s += 64) {
    loadPA(A1, s + 32);
    loadVB(B1, s + 32);
#pragma unroll
    for (int t = 0; t < 6; ++t)
      yacc[t] = __builtin_amdgcn_wmma_f32_16x16x32_f16(false, A0.v, false, B0[t].v,
                                                       (short)0, yacc[t], false, false);
    loadPA(A0, (s + 64) & (SS - 1));
    loadVB(B0, (s + 64) & (SS - 1));
#pragma unroll
    for (int t = 0; t < 6; ++t)
      yacc[t] = __builtin_amdgcn_wmma_f32_16x16x32_f16(false, A1.v, false, B1[t].v,
                                                       (short)0, yacc[t], false, false);
  }

  // ---- epilogue: 1/l scale, LayerNorm over D (cross-wave), w_out dot -------
  float s1[8], s2[8];
#pragma unroll
  for (int i = 0; i < 8; ++i) { s1[i] = 0.f; s2[i] = 0.f; }
#pragma unroll
  for (int i = 0; i < 8; ++i) {
    const float li = linv[i + lh * 8];
#pragma unroll
    for (int t = 0; t < 6; ++t) {
      float v = yacc[t][i] * li;
      yacc[t][i] = v;
      s1[i] += v; s2[i] += v * v;
    }
  }
#pragma unroll
  for (int i = 0; i < 8; ++i) {
    float a = halfsum16(s1[i]);
    float c = halfsum16(s2[i]);
    if (l == 0) { red1[wave * 16 + i + lh * 8] = a; red2[wave * 16 + i + lh * 8] = c; }
  }
  __syncthreads();
  if (tid < 16) {
    float a = 0.f, c = 0.f;
    for (int w = 0; w < 8; ++w) { a += red1[w * 16 + tid]; c += red2[w * 16 + tid]; }
    float mu  = a * (1.f / DD);
    float var = c * (1.f / DD) - mu * mu;
    muS[tid] = mu;
    rsS[tid] = rsqrtf(var + 1e-5f);
  }
  __syncthreads();

  float po[8];
#pragma unroll
  for (int i = 0; i < 8; ++i) po[i] = 0.f;
#pragma unroll
  for (int t = 0; t < 6; ++t) {
    const int d = n0 + t * 16 + l;
    const float gd = lng[d], bd = lnb[d], wd = wout[d];
#pragma unroll
    for (int i = 0; i < 8; ++i) {
      const int r = i + lh * 8;
      float v = (yacc[t][i] - muS[r]) * rsS[r] * gd + bd;
      po[i] += v * wd;
    }
  }
#pragma unroll
  for (int i = 0; i < 8; ++i) {
    float v = halfsum16(po[i]);
    if (l == 0) outred[wave * 16 + i + lh * 8] = v;
  }
  __syncthreads();
  if (tid < 16) {
    float o = bout[0];
    for (int w = 0; w < 8; ++w) o += outred[w * 16 + tid];
    out[(size_t)b * CC + c0 + tid] = o;
  }
}

// ---------------------------------------------------------------------------
extern "C" void kernel_launch(void* const* d_in, const int* in_sizes, int n_in,
                              void* d_out, int out_size, void* d_ws, size_t ws_size,
                              hipStream_t stream) {
  (void)in_sizes; (void)n_in; (void)out_size; (void)ws_size;
  const float* x  = (const float*)d_in[0];
  const int*   am = (const int*)d_in[1];
  const float* Wk = (const float*)d_in[2];
  const float* Wv = (const float*)d_in[3];
  const float* bv = (const float*)d_in[4];
  const float* Q  = (const float*)d_in[5];
  const float* lg = (const float*)d_in[6];
  const float* lb = (const float*)d_in[7];
  const float* wo = (const float*)d_in[8];
  const float* bo = (const float*)d_in[9];
  float* out = (float*)d_out;

  _Float16* Kh  = (_Float16*)d_ws;                      // [B][S][D] f16
  _Float16* Vt  = Kh  + (size_t)BB * SS * DD;           // [B][D][S] f16
  _Float16* WhK = Vt  + (size_t)BB * SS * DD;           // [D][D] f16
  _Float16* WhV = WhK + (size_t)DD * DD;                // [D][D] f16

  hipFuncSetAttribute(reinterpret_cast<const void*>(attn_kernel),
                      hipFuncAttributeMaxDynamicSharedMemorySize, (int)SMEM_BYTES);

  wcvt_kernel<<<dim3((DD * DD) / (256 * 4)), 256, 0, stream>>>(Wk, Wv, WhK, WhV);
  kv_proj_kernel<<<dim3((BB * SS) / 16, DD / 256), 256, 0, stream>>>(x, WhK, WhV, bv, Kh, Vt);
  attn_kernel<<<dim3(CC / 16, BB), 256, SMEM_BYTES, stream>>>(Q, am, Kh, Vt, lg, lb, wo, bo, out);
}
